// ProbSparseSelfAttention_27092653703470
// MI455X (gfx1250) — matrix operand order repaired and checked by reading, not compiled, not run on hardware
//
#include <hip/hip_runtime.h>
#include <hip/hip_bf16.h>

// ---------------------------------------------------------------------------
// ProbSparse self-attention for MI455X (gfx1250, wave32, WMMA).
// B=4, S=8192, D=512, H=8, hd=64, u = int(5*ln(8193)) = 45.
// ---------------------------------------------------------------------------

#define B_  4
#define S_  8192
#define D_  512
#define H_  8
#define HD  64
#define T_  (B_ * S_)     // 32768 tokens
#define U_  45            // active queries per (b,h)
#define UP  48            // padded to 3 WMMA M-tiles

typedef _Float16 half_t;
typedef _Float16 v16h __attribute__((ext_vector_type(16)));
typedef _Float16 v8h  __attribute__((ext_vector_type(8)));
typedef _Float16 v4h  __attribute__((ext_vector_type(4)));
typedef float    v8f  __attribute__((ext_vector_type(8)));

__device__ inline v8f zero8() {
  v8f z;
#pragma unroll
  for (int i = 0; i < 8; ++i) z[i] = 0.0f;
  return z;
}

__device__ inline v16h cat8(v8h lo, v8h hi) {
  return __builtin_shufflevector(lo, hi, 0, 1, 2, 3, 4, 5, 6, 7, 8, 9, 10, 11,
                                 12, 13, 14, 15);
}

// A-operand (16x32 f16, MxK), rowp = &A[row][0], K contiguous, 16B-aligned.
// lanes 0-15 : halves 0..7 = K0..7,  halves 8..15 = K16..23
// lanes 16-31: halves 0..7 = K8..15, halves 8..15 = K24..31
// -> two 16-byte vector loads (ds_load_b128 / global_load_b128).
__device__ inline v16h load_a16v(const half_t* rowp, int lane) {
  const int koff = (lane & 16) ? 8 : 0;
  v8h lo = *(const v8h*)(rowp + koff);
  v8h hi = *(const v8h*)(rowp + 16 + koff);
  return cat8(lo, hi);
}

// B-operand (32x16 f16, KxN). Lane l holds column N = l&15; colp = &B^T[col][k0].
// lanes 0-15 : halves 0..7 = K0..7,  halves 8..15 = K16..23
// lanes 16-31: halves 0..7 = K8..15, halves 8..15 = K24..31
// -> two 16-byte vector loads.
__device__ inline v16h load_b16v(const half_t* colp, int lane) {
  const int kb = (lane & 16) ? 8 : 0;
  v8h lo = *(const v8h*)(colp + kb);
  v8h hi = *(const v8h*)(colp + 16 + kb);
  return cat8(lo, hi);
}

__device__ inline v8f wmma_f16(v16h a, v16h b, v8f c) {
  return __builtin_amdgcn_wmma_f32_16x16x32_f16(false, a, false, b,
                                                (short)0, c, false, false);
}

// ---------------------------------------------------------------------------
// Kernel 1: fused QKV projection. grid (T/64, 1536/128), block 128 (4 waves).
// Each wave computes a 16x128 tile. For K/V: store f16 [B,H,S,hd].
// For Q: reduce sum(q^2) over each head's 64 columns -> sparsity metric.
// Double-buffered LDS; staging via float4 global loads + v4h DS stores
// (B tile transposed in-register as 4x4 micro-blocks).
// ---------------------------------------------------------------------------
__global__ __launch_bounds__(128) void qkv_kernel(
    const float* __restrict__ x,
    const float* __restrict__ Wq, const float* __restrict__ bq,
    const float* __restrict__ Wk, const float* __restrict__ bk,
    const float* __restrict__ Wv, const float* __restrict__ bv,
    half_t* __restrict__ kh, half_t* __restrict__ vh,
    float* __restrict__ sparsity) {
  __shared__ alignas(16) half_t As[2][64][40];    // [m][k], 80B row stride
  __shared__ alignas(16) half_t BsT[2][128][40];  // [n][k], 80B row stride

  const int tid  = threadIdx.x;
  const int lane = tid & 31;
  const int wave = tid >> 5;
  const int m0   = blockIdx.x * 64;
  const int n0   = blockIdx.y * 128;
  const int which = n0 >> 9;          // 0=Q 1=K 2=V (block never straddles)
  const int wn0   = n0 & 511;
  const float* W    = (which == 0) ? Wq : (which == 1) ? Wk : Wv;
  const float* bias = (which == 0) ? bq : (which == 1) ? bk : bv;

  // Stage one 64x32 A tile + 32x128 B tile (transposed) into buffer `buf`.
  auto stage = [&](int kk, int buf) {
    // A: 512 float4 loads across the block, 4 per thread, b64 DS stores.
#pragma unroll
    for (int i = 0; i < 4; ++i) {
      int l4 = i * 128 + tid;
      int r = l4 >> 3, kc = (l4 & 7) << 2;
      float4 f = *(const float4*)&x[(size_t)(m0 + r) * D_ + kk + kc];
      v4h h = {(half_t)f.x, (half_t)f.y, (half_t)f.z, (half_t)f.w};
      *(v4h*)&As[buf][r][kc] = h;
    }
    // B: 4x4 micro-blocks, 2 per thread; coalesced float4 reads along n,
    // in-register transpose, contiguous-in-K v4h stores.
#pragma unroll
    for (int i = 0; i < 2; ++i) {
      int bl = i * 128 + tid;            // 256 blocks: 8 (k) x 32 (n)
      int kb4 = (bl >> 5) << 2;
      int cb4 = (bl & 31) << 2;
      const float* wp = &W[(size_t)(kk + kb4) * D_ + wn0 + cb4];
      float4 f0 = *(const float4*)(wp + 0 * D_);
      float4 f1 = *(const float4*)(wp + 1 * D_);
      float4 f2 = *(const float4*)(wp + 2 * D_);
      float4 f3 = *(const float4*)(wp + 3 * D_);
      *(v4h*)&BsT[buf][cb4 + 0][kb4] =
          (v4h){(half_t)f0.x, (half_t)f1.x, (half_t)f2.x, (half_t)f3.x};
      *(v4h*)&BsT[buf][cb4 + 1][kb4] =
          (v4h){(half_t)f0.y, (half_t)f1.y, (half_t)f2.y, (half_t)f3.y};
      *(v4h*)&BsT[buf][cb4 + 2][kb4] =
          (v4h){(half_t)f0.z, (half_t)f1.z, (half_t)f2.z, (half_t)f3.z};
      *(v4h*)&BsT[buf][cb4 + 3][kb4] =
          (v4h){(half_t)f0.w, (half_t)f1.w, (half_t)f2.w, (half_t)f3.w};
    }
  };

  v8f acc[8];
#pragma unroll
  for (int i = 0; i < 8; ++i) acc[i] = zero8();

  stage(0, 0);
  int p = 0;
  for (int kk = 0; kk < D_; kk += 32) {
    __syncthreads();
    if (kk + 32 < D_) {
      stage(kk + 32, p ^ 1);            // overlaps with WMMA on buffer p
      __builtin_prefetch(&x[(size_t)(m0 + (tid >> 1)) * D_ + kk + 64], 0, 1);
    }
    const int arow = wave * 16 + (lane & 15);
    v16h a = load_a16v(&As[p][arow][0], lane);
#pragma unroll
    for (int nt = 0; nt < 8; ++nt) {
      v16h b = load_b16v(&BsT[p][nt * 16 + (lane & 15)][0], lane);
      acc[nt] = wmma_f16(a, b, acc[nt]);
    }
    p ^= 1;
  }

  // Epilogue. C layout: VGPR i <-> (M = i + 8*(lane>=16), N = lane&15).
  const int ncol15 = lane & 15;
  const int mbase  = m0 + wave * 16 + ((lane >> 4) << 3);
  float ssum[2][8];
#pragma unroll
  for (int g = 0; g < 2; ++g)
#pragma unroll
    for (int i = 0; i < 8; ++i) ssum[g][i] = 0.0f;

#pragma unroll
  for (int nt = 0; nt < 8; ++nt) {
    const int wn = wn0 + nt * 16 + ncol15;       // column inside this W
    const float bval = bias[wn];
    const int h = wn >> 6, d = wn & 63;
#pragma unroll
    for (int i = 0; i < 8; ++i) {
      float val = acc[nt][i] + bval;
      int srow = mbase + i;
      int b = srow >> 13, s = srow & (S_ - 1);
      size_t o = ((size_t)(b * H_ + h) * S_ + s) * HD + d;
      if (which == 1)       kh[o] = (half_t)val;
      else if (which == 2)  vh[o] = (half_t)val;
      else                  ssum[nt >> 2][i] += val * val;  // Q: ||q||^2
    }
  }
  if (which == 0) {
#pragma unroll
    for (int g = 0; g < 2; ++g) {
      const int h = (wn0 + g * 64) >> 6;
#pragma unroll
      for (int i = 0; i < 8; ++i) {
        float v = ssum[g][i];
        v += __shfl_xor(v, 1, 16);      // reduce over 16 N-lanes only
        v += __shfl_xor(v, 2, 16);
        v += __shfl_xor(v, 4, 16);
        v += __shfl_xor(v, 8, 16);
        if (ncol15 == 0) {
          int srow = mbase + i;
          int b = srow >> 13, s = srow & (S_ - 1);
          sparsity[(size_t)(b * H_ + h) * S_ + s] = v;
        }
      }
    }
  }
}

// ---------------------------------------------------------------------------
// Kernel 2: V transpose [B,H,S,hd] -> [B,H,hd,S] (for WMMA B-operand).
// grid = B*H*(S/64), block 256.
// ---------------------------------------------------------------------------
__global__ __launch_bounds__(256) void transpose_v_kernel(
    const half_t* __restrict__ vh, half_t* __restrict__ vt) {
  __shared__ alignas(16) half_t tile[64][72];
  const int g = blockIdx.x;
  const int schunk = g & (S_ / 64 - 1);
  const int bh = g / (S_ / 64);
  const int tid = threadIdx.x;
  const half_t* src = vh + (size_t)bh * S_ * HD + (size_t)schunk * 64 * HD;
#pragma unroll
  for (int i = 0; i < 2; ++i) {        // v8h loads: 512 per tile, 2/thread
    int l8 = i * 256 + tid;
    int r = l8 >> 3, c = (l8 & 7) << 3;
    *(v8h*)&tile[r][c] = *(const v8h*)&src[r * HD + c];
  }
  __syncthreads();
  half_t* dst = vt + (size_t)bh * HD * S_ + schunk * 64;
#pragma unroll
  for (int i = 0; i < 16; ++i) {
    int l = i * 256 + tid;
    int d = l >> 6, s = l & 63;
    dst[(size_t)d * S_ + s] = tile[s][d];
  }
}

// ---------------------------------------------------------------------------
// Kernel 3: top-45 per (b,h) via iterative argmax on an LDS copy. grid 32.
// ---------------------------------------------------------------------------
__global__ __launch_bounds__(256) void topk_kernel(
    const float* __restrict__ sparsity, int* __restrict__ topidx) {
  __shared__ float vals[S_];     // 32 KB
  __shared__ float rmax[256];
  __shared__ int   ridx[256];
  const int bh = blockIdx.x;
  const int tid = threadIdx.x;
  const float* sp = sparsity + (size_t)bh * S_;
  for (int i = tid; i < S_; i += 256) vals[i] = sp[i];
  __syncthreads();
  for (int it = 0; it < U_; ++it) {
    float best = -3.4e38f; int bi = 0;
    for (int i = tid; i < S_; i += 256) {
      float v = vals[i];
      if (v > best) { best = v; bi = i; }
    }
    rmax[tid] = best; ridx[tid] = bi;
    __syncthreads();
    for (int off = 128; off > 0; off >>= 1) {
      if (tid < off && rmax[tid + off] > rmax[tid]) {
        rmax[tid] = rmax[tid + off]; ridx[tid] = ridx[tid + off];
      }
      __syncthreads();
    }
    if (tid == 0) { topidx[bh * UP + it] = ridx[0]; vals[ridx[0]] = -3.4e38f; }
    __syncthreads();
  }
  if (tid < UP - U_) topidx[bh * UP + U_ + tid] = 0;   // padding rows
}

// ---------------------------------------------------------------------------
// Kernel 4: gather/recompute selected queries in fp32 (45 rows/head, tiny),
// rounded to f16; also zero selctx accumulator. grid 32.
// ---------------------------------------------------------------------------
__global__ __launch_bounds__(256) void selq_kernel(
    const float* __restrict__ x, const float* __restrict__ Wq,
    const float* __restrict__ bq, const int* __restrict__ topidx,
    half_t* __restrict__ selq, float* __restrict__ selctx) {
  const int bh = blockIdx.x;
  const int b = bh >> 3, h = bh & 7;
  for (int e = threadIdx.x; e < UP * HD; e += 256) {
    int j = e >> 6, d = e & 63;
    float acc = 0.0f;
    if (j < U_) {
      int s = topidx[bh * UP + j];
      const float* xr = x + ((size_t)b * S_ + s) * D_;
      int n = h * HD + d;
      acc = bq[n];
      for (int k = 0; k < D_; ++k) acc += xr[k] * Wq[(size_t)k * D_ + n];
    }
    selq[(size_t)bh * UP * HD + e]   = (half_t)acc;
    selctx[(size_t)bh * UP * HD + e] = 0.0f;
  }
}

// ---------------------------------------------------------------------------
// Kernel 5: scores = selq @ K^T * hd^-0.5 via WMMA, stored f16 [UP][S].
// grid (32, 8), block 128 (4 waves). B-operand = contiguous K rows (b128).
// ---------------------------------------------------------------------------
__global__ __launch_bounds__(128) void scores_kernel(
    const half_t* __restrict__ kh, const half_t* __restrict__ selq,
    half_t* __restrict__ attn) {
  __shared__ alignas(16) half_t sq[UP][HD];
  const int bh = blockIdx.x, sg = blockIdx.y;
  const int tid = threadIdx.x, lane = tid & 31, wave = tid >> 5;
  const half_t* sqg = selq + (size_t)bh * UP * HD;
  for (int i = tid; i < (UP * HD) / 8; i += 128)
    *(v8h*)&sq[0][i * 8] = *(const v8h*)&sqg[i * 8];
  __syncthreads();

  v16h afr[3][2];
#pragma unroll
  for (int mt = 0; mt < 3; ++mt)
#pragma unroll
    for (int kx = 0; kx < 2; ++kx)
      afr[mt][kx] = load_a16v(&sq[mt * 16 + (lane & 15)][kx * 32], lane);

  const half_t* kbase = kh + (size_t)bh * S_ * HD;
  half_t* arow = attn + (size_t)bh * UP * S_;

  for (int ch = wave; ch < 64; ch += 4) {
    const int s0 = sg * 1024 + ch * 16;
    v8f acc[3] = {zero8(), zero8(), zero8()};
#pragma unroll
    for (int kx = 0; kx < 2; ++kx) {
      v16h bfr = load_b16v(kbase + (size_t)(s0 + (lane & 15)) * HD + kx * 32,
                           lane);
#pragma unroll
      for (int mt = 0; mt < 3; ++mt) acc[mt] = wmma_f16(afr[mt][kx], bfr, acc[mt]);
    }
    const int scol = s0 + (lane & 15);
#pragma unroll
    for (int mt = 0; mt < 3; ++mt)
#pragma unroll
      for (int i = 0; i < 8; ++i) {
        int j = mt * 16 + ((lane >> 4) << 3) + i;
        arow[(size_t)j * S_ + scol] = (half_t)(acc[mt][i] * 0.125f); // hd^-0.5
      }
  }
}

// ---------------------------------------------------------------------------
// Kernel 6: exact row softmax over the 45 real rows. grid (32, 45).
// ---------------------------------------------------------------------------
__global__ __launch_bounds__(256) void softmax_kernel(half_t* __restrict__ attn) {
  const int bh = blockIdx.x, j = blockIdx.y;
  half_t* row = attn + ((size_t)bh * UP + j) * S_;
  __shared__ float red[256];
  const int tid = threadIdx.x;
  float m = -3.4e38f;
  for (int i = tid; i < S_; i += 256) m = fmaxf(m, (float)row[i]);
  red[tid] = m; __syncthreads();
  for (int off = 128; off > 0; off >>= 1) {
    if (tid < off) red[tid] = fmaxf(red[tid], red[tid + off]);
    __syncthreads();
  }
  m = red[0]; __syncthreads();
  float ssum = 0.0f;
  for (int i = tid; i < S_; i += 256) ssum += expf((float)row[i] - m);
  red[tid] = ssum; __syncthreads();
  for (int off = 128; off > 0; off >>= 1) {
    if (tid < off) red[tid] += red[tid + off];
    __syncthreads();
  }
  const float inv = 1.0f / red[0];
  __syncthreads();
  for (int i = tid; i < S_; i += 256)
    row[i] = (half_t)(expf((float)row[i] - m) * inv);
}

// ---------------------------------------------------------------------------
// Kernel 7: ctx = attn @ V via WMMA (K=8192 split over grid.y and waves),
// LDS ds_add_f32 reduction then global f32 atomics. grid (32, 8), block 128.
// Both operands load as b128 pairs (attn rows, V^T rows).
// ---------------------------------------------------------------------------
__global__ __launch_bounds__(128) void ctx_kernel(
    const half_t* __restrict__ attn, const half_t* __restrict__ vt,
    float* __restrict__ selctx) {
  __shared__ float cacc[UP][HD];   // 12 KB
  const int bh = blockIdx.x, kg = blockIdx.y;
  const int tid = threadIdx.x, lane = tid & 31, wave = tid >> 5;
  for (int i = tid; i < UP * HD; i += 128) (&cacc[0][0])[i] = 0.0f;
  __syncthreads();

  const half_t* ab = attn + (size_t)bh * UP * S_;
  const half_t* vb = vt + (size_t)bh * HD * S_;
  v8f acc[3][4];
#pragma unroll
  for (int mt = 0; mt < 3; ++mt)
#pragma unroll
    for (int nt = 0; nt < 4; ++nt) acc[mt][nt] = zero8();

  for (int ks = wave; ks < 32; ks += 4) {
    const int k0 = kg * 1024 + ks * 32;
    v16h afr[3], bfr[4];
#pragma unroll
    for (int mt = 0; mt < 3; ++mt)
      afr[mt] = load_a16v(ab + (size_t)(mt * 16 + (lane & 15)) * S_ + k0, lane);
#pragma unroll
    for (int nt = 0; nt < 4; ++nt)
      bfr[nt] = load_b16v(vb + (size_t)(nt * 16 + (lane & 15)) * S_ + k0, lane);
#pragma unroll
    for (int mt = 0; mt < 3; ++mt)
#pragma unroll
      for (int nt = 0; nt < 4; ++nt)
        acc[mt][nt] = wmma_f16(afr[mt], bfr[nt], acc[mt][nt]);
  }

#pragma unroll
  for (int mt = 0; mt < 3; ++mt)
#pragma unroll
    for (int nt = 0; nt < 4; ++nt)
#pragma unroll
      for (int i = 0; i < 8; ++i) {
        int j = mt * 16 + ((lane >> 4) << 3) + i;
        int d = nt * 16 + (lane & 15);
        atomicAdd(&cacc[j][d], acc[mt][nt][i]);     // ds_add_f32
      }
  __syncthreads();
  float* sc = selctx + (size_t)bh * UP * HD;
  for (int i = tid; i < UP * HD; i += 128) atomicAdd(&sc[i], (&cacc[0][0])[i]);
}

// ---------------------------------------------------------------------------
// Kernel 8: vmean over S (from V^T rows, contiguous). grid 32.
// ---------------------------------------------------------------------------
__global__ __launch_bounds__(256) void vmean_kernel(
    const half_t* __restrict__ vt, float* __restrict__ vmean) {
  __shared__ float part[256];
  const int bh = blockIdx.x;
  const int tid = threadIdx.x;
  const int d = tid >> 2, p = tid & 3;
  const half_t* row = vt + ((size_t)bh * HD + d) * S_;
  float s = 0.0f;
  for (int i = p * 2048; i < (p + 1) * 2048; ++i) s += (float)row[i];
  part[tid] = s;
  __syncthreads();
  if (p == 0)
    vmean[bh * HD + d] =
        (part[tid] + part[tid + 1] + part[tid + 2] + part[tid + 3]) *
        (1.0f / S_);
}

// ---------------------------------------------------------------------------
// Kernel 9: base output row per batch: concat_h(vmean) @ Wo + bo. grid 4.
// ---------------------------------------------------------------------------
__global__ __launch_bounds__(512) void baseout_kernel(
    const float* __restrict__ vmean, const float* __restrict__ Wo,
    const float* __restrict__ bo, float* __restrict__ baseout) {
  const int b = blockIdx.x, n = threadIdx.x;
  float acc = bo[n];
  for (int k = 0; k < D_; ++k) {
    int h = k >> 6, d = k & 63;
    acc += vmean[(b * H_ + h) * HD + d] * Wo[(size_t)k * D_ + n];
  }
  baseout[b * D_ + n] = acc;
}

// ---------------------------------------------------------------------------
// Kernel 10: broadcast base row to all tokens (float4). grid 16384.
// ---------------------------------------------------------------------------
__global__ __launch_bounds__(256) void broadcast_kernel(
    const float* __restrict__ baseout, float* __restrict__ out) {
  const size_t idx = (size_t)blockIdx.x * 256 + threadIdx.x;  // float4 index
  const int n4 = (int)(idx & 127);
  const size_t row = idx >> 7;
  const int b = (int)(row >> 13);
  ((float4*)out)[idx] = ((const float4*)baseout)[b * 128 + n4];
}

// ---------------------------------------------------------------------------
// Kernel 11: rank-64 delta update for selected rows:
// out[b,s,:] += (selctx - vmean_h) @ Wo[h*64:(h+1)*64, :]. grid 32*45.
// ---------------------------------------------------------------------------
__global__ __launch_bounds__(256) void scatter_kernel(
    const float* __restrict__ selctx, const float* __restrict__ vmean,
    const int* __restrict__ topidx, const float* __restrict__ Wo,
    float* __restrict__ out) {
  __shared__ float delta[HD];
  const int g = blockIdx.x;
  const int bh = g / U_;
  const int j = g - bh * U_;
  const int b = bh >> 3, h = bh & 7;
  const int tid = threadIdx.x;
  if (tid < HD)
    delta[tid] = selctx[((size_t)bh * UP + j) * HD + tid] - vmean[bh * HD + tid];
  __syncthreads();
  const int s = topidx[bh * UP + j];
  float* orow = out + ((size_t)b * S_ + s) * D_;
  for (int n = tid; n < D_; n += 256) {
    float acc = 0.0f;
#pragma unroll
    for (int d = 0; d < HD; ++d) acc += delta[d] * Wo[(size_t)(h * HD + d) * D_ + n];
    atomicAdd(&orow[n], acc);
  }
}

// ---------------------------------------------------------------------------
// Host launcher. Workspace layout (bytes), total ~122 MB.
// ---------------------------------------------------------------------------
extern "C" void kernel_launch(void* const* d_in, const int* in_sizes, int n_in,
                              void* d_out, int out_size, void* d_ws,
                              size_t ws_size, hipStream_t stream) {
  (void)in_sizes; (void)n_in; (void)out_size; (void)ws_size;
  const float* x  = (const float*)d_in[0];
  const float* Wq = (const float*)d_in[1];
  const float* bq = (const float*)d_in[2];
  const float* Wk = (const float*)d_in[3];
  const float* bk = (const float*)d_in[4];
  const float* Wv = (const float*)d_in[5];
  const float* bv = (const float*)d_in[6];
  const float* Wo = (const float*)d_in[7];
  const float* bo = (const float*)d_in[8];
  float* out = (float*)d_out;

  char* ws = (char*)d_ws;
  const size_t SZ_KH    = (size_t)B_ * H_ * S_ * HD * sizeof(half_t); // 32 MB
  const size_t SZ_ATTN  = (size_t)B_ * H_ * UP * S_ * sizeof(half_t); // 24 MB
  const size_t SZ_SPARS = (size_t)B_ * H_ * S_ * sizeof(float);       // 1 MB
  size_t off = 0;
  half_t* kh     = (half_t*)(ws + off); off += SZ_KH;
  half_t* vh     = (half_t*)(ws + off); off += SZ_KH;
  half_t* vt     = (half_t*)(ws + off); off += SZ_KH;
  half_t* attn   = (half_t*)(ws + off); off += SZ_ATTN;
  float*  spars  = (float*)(ws + off);  off += SZ_SPARS;
  int*    topidx = (int*)(ws + off);    off += (size_t)B_ * H_ * UP * sizeof(int);
  off = (off + 255) & ~(size_t)255;
  half_t* selq   = (half_t*)(ws + off); off += (size_t)B_ * H_ * UP * HD * sizeof(half_t);
  off = (off + 255) & ~(size_t)255;
  float*  selctx = (float*)(ws + off);  off += (size_t)B_ * H_ * UP * HD * sizeof(float);
  float*  vmean  = (float*)(ws + off);  off += (size_t)B_ * H_ * HD * sizeof(float);
  float*  baseo  = (float*)(ws + off);  off += (size_t)B_ * D_ * sizeof(float);

  qkv_kernel<<<dim3(T_ / 64, (3 * D_) / 128), 128, 0, stream>>>(
      x, Wq, bq, Wk, bk, Wv, bv, kh, vh, spars);
  transpose_v_kernel<<<B_ * H_ * (S_ / 64), 256, 0, stream>>>(vh, vt);
  topk_kernel<<<B_ * H_, 256, 0, stream>>>(spars, topidx);
  selq_kernel<<<B_ * H_, 256, 0, stream>>>(x, Wq, bq, topidx, selq, selctx);
  scores_kernel<<<dim3(B_ * H_, 8), 128, 0, stream>>>(kh, selq, attn);
  softmax_kernel<<<dim3(B_ * H_, U_), 256, 0, stream>>>(attn);
  ctx_kernel<<<dim3(B_ * H_, 8), 128, 0, stream>>>(attn, vt, selctx);
  vmean_kernel<<<B_ * H_, 256, 0, stream>>>(vt, vmean);
  baseout_kernel<<<B_, 512, 0, stream>>>(vmean, Wo, bo, baseo);
  broadcast_kernel<<<(B_ * S_ * D_ / 4) / 256, 256, 0, stream>>>(baseo, out);
  scatter_kernel<<<B_ * H_ * U_, 256, 0, stream>>>(selctx, vmean, topidx, Wo, out);
}
